// EGCL_27986006901442
// MI455X (gfx1250) — matrix operand order repaired and checked by reading, not compiled, not run on hardware
//
#include <hip/hip_runtime.h>
#include <hip/hip_bf16.h>
#include <stdint.h>

#define NN 50000
#define NE 640000
#define RAW 128
#define HID 256

typedef __attribute__((ext_vector_type(16))) __bf16 v16bf;
typedef __attribute__((ext_vector_type(8)))  float  v8f;

union Frag { v16bf v; uint4 q[2]; unsigned short u[16]; };

__device__ __forceinline__ unsigned short f2bf(float f) {
    unsigned int u = __float_as_uint(f);
    u += 0x7fffu + ((u >> 16) & 1u);          // round-to-nearest-even
    return (unsigned short)(u >> 16);
}
__device__ __forceinline__ float bf2f(unsigned short s) {
    return __uint_as_float(((unsigned int)s) << 16);
}

// fast SiLU: x * sigmoid(x), sigmoid via v_exp_f32 + v_rcp_f32 (no IEEE divide)
__device__ __forceinline__ float silu(float x) {
#if __has_builtin(__builtin_amdgcn_rcpf)
    return x * __builtin_amdgcn_rcpf(1.f + __expf(-x));
#else
    return x / (1.f + __expf(-x));
#endif
}
__device__ __forceinline__ void atomAddF(float* p, float v) {
    (void)__hip_atomic_fetch_add(p, v, __ATOMIC_RELAXED, __HIP_MEMORY_SCOPE_AGENT);
}

// pack two f32 -> packed bf16x2 (low = a, high = b)
#if __has_builtin(__builtin_amdgcn_cvt_pk_bf16_f32)
typedef __attribute__((ext_vector_type(2))) __bf16 v2bf;
__device__ __forceinline__ unsigned int pack2(float a, float b) {
    union { v2bf v; unsigned int u; } cv;
    cv.v = __builtin_amdgcn_cvt_pk_bf16_f32(a, b);
    return cv.u;
}
#else
__device__ __forceinline__ unsigned int pack2(float a, float b) {
    return (unsigned int)f2bf(a) | ((unsigned int)f2bf(b) << 16);
}
#endif

// 16xN GEMM stage: A (16 rows) staged in LDS as bf16 [16][stride],
// B tiles pre-swizzled in global as [KC][NT][512] bf16 in fragment order
// (lane*16 + slot). KC = number of K=32 chunks, NT = number of N=16 tiles.
// A fragment layout (16-bit A 16x32, wave32):
//   lanes 0-15 : M=lane,    slots 0-7 = K kb+0..7, slots 8-15 = K kb+16..23, kb=0
//   lanes 16-31: M=lane-16, kb=8
template<int KC, int NT, typename Epi>
__device__ __forceinline__ void mm_stage(const unsigned short* lds_in, int stride,
                                         const unsigned short* __restrict__ wt,
                                         int lane, Epi epi)
{
    const int row = lane & 15, half = lane >> 4;
    Frag a[KC];
    const unsigned short* ab = lds_in + row * stride + half * 8;
#pragma unroll
    for (int kc = 0; kc < KC; kc++) {
        a[kc].q[0] = *(const uint4*)(ab + kc * 32);        // K kb..kb+7
        a[kc].q[1] = *(const uint4*)(ab + kc * 32 + 16);   // K kb+16..kb+23
    }
#pragma unroll
    for (int nt = 0; nt < NT; nt++) {
        v8f acc = {};
        const unsigned short* bt = wt + nt * 512 + lane * 16;
#pragma unroll
        for (int kc = 0; kc < KC; kc++) {
            Frag b;
            b.q[0] = *(const uint4*)(bt + (size_t)kc * NT * 512);
            b.q[1] = *(const uint4*)(bt + (size_t)kc * NT * 512 + 8);
            acc = __builtin_amdgcn_wmma_f32_16x16x32_bf16(
                false, a[kc].v, false, b.v, (short)0, acc, false, false);
        }
        epi(nt, acc);
    }
}

// ---- weight prep: f32 row-major [K][ncols] -> bf16 B-fragment tiles ----
// B fragment (16-bit B 32x16, wave32): lane = half*16 + n_local, slot s:
// k_local = 8*half + s (s<8), 16 + 8*half + (s-8) (s>=8).  Stored linearly: tile[kc][nt][lane*16 + slot].
__global__ void prep_w(const float* __restrict__ src, unsigned short* __restrict__ dst,
                       int KC, int NT, int ncols)
{
    int idx = blockIdx.x * blockDim.x + threadIdx.x;
    int total = KC * NT * 512;
    if (idx >= total) return;
    int kc = idx / (NT * 512);
    int r  = idx - kc * NT * 512;
    int nt = r >> 9;
    int q  = r & 511;
    int lane = q >> 4, slot = q & 15;
    int k = kc * 32 + ((slot >> 3) << 4) + ((lane >> 4) << 3) + (slot & 7);
    int n = nt * 16 + (lane & 15);
    dst[idx] = f2bf(src[(size_t)k * ncols + n]);
}

__global__ void zero_ws(float* __restrict__ p, size_t n)
{
    size_t i = (size_t)blockIdx.x * blockDim.x + threadIdx.x;
    size_t stride = (size_t)gridDim.x * blockDim.x;
    for (; i < n; i += stride) p[i] = 0.f;
}

// ---- fused edge kernel: one wave per 16 edges ----
__global__ __launch_bounds__(32) void egcl_edge(
    const int*   __restrict__ ei,      const float* __restrict__ strf,
    const float* __restrict__ coordf,
    const float* __restrict__ msg_w1,  const float* __restrict__ msg_b1,
    const float* __restrict__ msg_b2,
    const float* __restrict__ tr_b1,   const float* __restrict__ tr_b2,
    const float* __restrict__ tr_w3,
    const unsigned short* __restrict__ W1, const unsigned short* __restrict__ W2,
    const unsigned short* __restrict__ T1, const unsigned short* __restrict__ T2,
    float* __restrict__ msg_sum, float* __restrict__ trans_sum, float* __restrict__ cnt)
{
    __shared__ unsigned short sIn[16 * 256];
    __shared__ unsigned short sA[16 * 256];
    __shared__ unsigned short sB[16 * 256];
    __shared__ int   sIdx[32];
    __shared__ float sDist[16];
    __shared__ float sCD[48];

    const int lane = threadIdx.x;
    const int half = lane >> 4, l = lane & 15;
    const int e0 = blockIdx.x * 16;

    // warm WGP$ for the weight tiles this wave will stream 4x16x8 times
    __builtin_prefetch(W1 + lane * 16, 0, 3);
    __builtin_prefetch(W2 + lane * 16, 0, 3);
    __builtin_prefetch(T1 + lane * 16, 0, 3);
    __builtin_prefetch(T2 + lane * 16, 0, 3);

    sIdx[lane] = ei[half * NE + e0 + l];   // 0-15: row, 16-31: col
    __syncthreads();

    if (half == 0) {
        int r = sIdx[l], c = sIdx[16 + l];
        float dx = coordf[r * 3 + 0] - coordf[c * 3 + 0];
        float dy = coordf[r * 3 + 1] - coordf[c * 3 + 1];
        float dz = coordf[r * 3 + 2] - coordf[c * 3 + 2];
        sCD[l * 3 + 0] = dx; sCD[l * 3 + 1] = dy; sCD[l * 3 + 2] = dz;
        sDist[l] = dx * dx + dy * dy + dz * dz;
    }

    // gather str[row] | str[col] -> bf16 edge_in [16][256]
#pragma unroll 4
    for (int m = 0; m < 16; m++) {
        int node = sIdx[half * 16 + m];
        const float4* src = (const float4*)(strf + (size_t)node * RAW) + (l << 1);
        float4 p = src[0], q = src[1];
        uint4 pk;
        pk.x = pack2(p.x, p.y); pk.y = pack2(p.z, p.w);
        pk.z = pack2(q.x, q.y); pk.w = pack2(q.z, q.w);
        *(uint4*)&sIn[m * 256 + half * 128 + l * 8] = pk;
    }
    __syncthreads();

    const float* w1d = msg_w1 + 256 * 256;   // dist row of msg_w1 (rank-1 update)

    // h1 = silu(edge_in @ w1 + b1)
    mm_stage<8, 16>(sIn, 256, W1, lane, [&](int nt, v8f acc) {
#pragma unroll
        for (int v = 0; v < 8; v++) {
            int m = v + half * 8, n = nt * 16 + l;
            float x = acc[v] + sDist[m] * w1d[n] + msg_b1[n];
            sA[m * 256 + n] = f2bf(silu(x));
        }
    });
    __syncthreads();

    // msg = silu(h1 @ w2 + b2);  msg_sum[row] += msg (f32 atomics from accumulators)
    mm_stage<8, 16>(sA, 256, W2, lane, [&](int nt, v8f acc) {
#pragma unroll
        for (int v = 0; v < 8; v++) {
            int m = v + half * 8, n = nt * 16 + l;
            float h = silu(acc[v] + msg_b2[n]);
            atomAddF(&msg_sum[(size_t)sIdx[m] * HID + n], h);
            sB[m * 256 + n] = f2bf(h);
        }
    });
    __syncthreads();

    // t1 = silu(msg @ tr_w1 + tr_b1)
    mm_stage<8, 16>(sB, 256, T1, lane, [&](int nt, v8f acc) {
#pragma unroll
        for (int v = 0; v < 8; v++) {
            int m = v + half * 8, n = nt * 16 + l;
            sA[m * 256 + n] = f2bf(silu(acc[v] + tr_b1[n]));
        }
    });
    __syncthreads();

    // t2 = silu(t1 @ tr_w2 + tr_b2)
    mm_stage<8, 16>(sA, 256, T2, lane, [&](int nt, v8f acc) {
#pragma unroll
        for (int v = 0; v < 8; v++) {
            int m = v + half * 8, n = nt * 16 + l;
            sB[m * 256 + n] = f2bf(silu(acc[v] + tr_b2[n]));
        }
    });
    __syncthreads();

    // gate = t2 @ tr_w3 : both half-waves each do 128 K, combine via lane permute
    {
        float g = 0.f;
        const unsigned short* t2 = &sB[l * 256 + half * 128];
        const float* w3 = tr_w3 + half * 128;
#pragma unroll 8
        for (int k = 0; k < 128; k++) g += bf2f(t2[k]) * w3[k];
        g += __shfl_xor(g, 16, 32);   // lanes 0-15 now hold the full dot
        if (half == 0) {
            int r = sIdx[l];
            atomAddF(&trans_sum[(size_t)r * 3 + 0], sCD[l * 3 + 0] * g);
            atomAddF(&trans_sum[(size_t)r * 3 + 1], sCD[l * 3 + 1] * g);
            atomAddF(&trans_sum[(size_t)r * 3 + 2], sCD[l * 3 + 2] * g);
            atomAddF(&cnt[r], 1.f);
        }
    }
}

// ---- node kernel: posi_mlp + residual + coord update, one wave per 16 nodes ----
__global__ __launch_bounds__(32) void egcl_node(
    const float* __restrict__ strf, const float* __restrict__ coordf,
    const float* __restrict__ msg_sum,
    const float* __restrict__ posi_b1, const float* __restrict__ posi_b2,
    const unsigned short* __restrict__ P1, const unsigned short* __restrict__ P2,
    const float* __restrict__ trans_sum, const float* __restrict__ cnt,
    float* __restrict__ out_str, float* __restrict__ out_coord)
{
    __shared__ unsigned short sIn[16 * 384];
    __shared__ unsigned short sH[16 * 256];

    const int lane = threadIdx.x;
    const int half = lane >> 4, l = lane & 15;
    const int n0 = blockIdx.x * 16;

    __builtin_prefetch(P1 + lane * 16, 0, 3);
    __builtin_prefetch(P2 + lane * 16, 0, 3);

    // stage [str | msg_sum] -> bf16 [16][384]; 96 float4 per node, 3 per lane
#pragma unroll 2
    for (int m = 0; m < 16; m++) {
        int node = n0 + m;
#pragma unroll
        for (int j = 0; j < 3; j++) {
            int i4 = lane * 3 + j;   // 0..95
            float4 p;
            if (i4 < 32) p = ((const float4*)(strf + (size_t)node * RAW))[i4];
            else         p = ((const float4*)(msg_sum + (size_t)node * HID))[i4 - 32];
            uint2 pk; pk.x = pack2(p.x, p.y); pk.y = pack2(p.z, p.w);
            *(uint2*)&sIn[m * 384 + i4 * 4] = pk;
        }
    }
    __syncthreads();

    // h = silu(p_in @ posi_w1 + posi_b1)
    mm_stage<12, 16>(sIn, 384, P1, lane, [&](int nt, v8f acc) {
#pragma unroll
        for (int v = 0; v < 8; v++) {
            int m = v + half * 8, n = nt * 16 + l;
            sH[m * 256 + n] = f2bf(silu(acc[v] + posi_b1[n]));
        }
    });
    __syncthreads();

    // str_out = str + h @ posi_w2 + posi_b2
    mm_stage<8, 8>(sH, 256, P2, lane, [&](int nt, v8f acc) {
#pragma unroll
        for (int v = 0; v < 8; v++) {
            int m = v + half * 8, n = nt * 16 + l;
            size_t o = (size_t)(n0 + m) * RAW + n;
            out_str[o] = strf[o] + acc[v] + posi_b2[n];
        }
    });

    // coord_out = coord + trans_sum / max(cnt, 1)
    if (half == 0) {
        int node = n0 + l;
        float d = fmaxf(cnt[node], 1.f);
#pragma unroll
        for (int k = 0; k < 3; k++)
            out_coord[(size_t)node * 3 + k] =
                coordf[(size_t)node * 3 + k] + trans_sum[(size_t)node * 3 + k] / d;
    }
}

extern "C" void kernel_launch(void* const* d_in, const int* in_sizes, int n_in,
                              void* d_out, int out_size, void* d_ws, size_t ws_size,
                              hipStream_t stream)
{
    const int*   ei      = (const int*)d_in[0];
    const float* strf    = (const float*)d_in[1];
    const float* coordf  = (const float*)d_in[2];
    const float* msg_w1  = (const float*)d_in[3];
    const float* msg_b1  = (const float*)d_in[4];
    const float* msg_w2  = (const float*)d_in[5];
    const float* msg_b2  = (const float*)d_in[6];
    const float* tr_w1   = (const float*)d_in[7];
    const float* tr_b1   = (const float*)d_in[8];
    const float* tr_w2   = (const float*)d_in[9];
    const float* tr_b2   = (const float*)d_in[10];
    const float* tr_w3   = (const float*)d_in[11];
    const float* posi_w1 = (const float*)d_in[12];
    const float* posi_b1 = (const float*)d_in[13];
    const float* posi_w2 = (const float*)d_in[14];
    const float* posi_b2 = (const float*)d_in[15];
    (void)in_sizes; (void)n_in; (void)out_size; (void)ws_size;

    // workspace layout (f32 accumulators first, then bf16 weight tiles)
    float* wsf       = (float*)d_ws;
    float* msg_sum   = wsf;                               // NN*256
    float* trans_sum = msg_sum + (size_t)NN * HID;        // NN*3
    float* cnt       = trans_sum + (size_t)NN * 3;        // NN
    unsigned short* W1 = (unsigned short*)(cnt + NN);     // 16B-aligned (13e6 floats)
    unsigned short* W2 = W1 + 8 * 16 * 512;
    unsigned short* T1 = W2 + 8 * 16 * 512;
    unsigned short* T2 = T1 + 8 * 16 * 512;
    unsigned short* P1 = T2 + 8 * 16 * 512;
    unsigned short* P2 = P1 + 12 * 16 * 512;

    size_t nz = (size_t)NN * HID + (size_t)NN * 3 + NN;
    zero_ws<<<2048, 256, 0, stream>>>(wsf, nz);

    prep_w<<<(8  * 16 * 512 + 255) / 256, 256, 0, stream>>>(msg_w1,  W1, 8, 16, 256);
    prep_w<<<(8  * 16 * 512 + 255) / 256, 256, 0, stream>>>(msg_w2,  W2, 8, 16, 256);
    prep_w<<<(8  * 16 * 512 + 255) / 256, 256, 0, stream>>>(tr_w1,   T1, 8, 16, 256);
    prep_w<<<(8  * 16 * 512 + 255) / 256, 256, 0, stream>>>(tr_w2,   T2, 8, 16, 256);
    prep_w<<<(12 * 16 * 512 + 255) / 256, 256, 0, stream>>>(posi_w1, P1, 12, 16, 256);
    prep_w<<<(8  * 8  * 512 + 255) / 256, 256, 0, stream>>>(posi_w2, P2, 8, 8, 128);

    egcl_edge<<<NE / 16, 32, 0, stream>>>(ei, strf, coordf, msg_w1, msg_b1, msg_b2,
                                          tr_b1, tr_b2, tr_w3, W1, W2, T1, T2,
                                          msg_sum, trans_sum, cnt);

    float* out_str   = (float*)d_out;
    float* out_coord = out_str + (size_t)NN * RAW;
    egcl_node<<<NN / 16, 32, 0, stream>>>(strf, coordf, msg_sum, posi_b1, posi_b2,
                                          P1, P2, trans_sum, cnt, out_str, out_coord);
}
